// GPT_14310831030714
// MI455X (gfx1250) — matrix operand order repaired and checked by reading, not compiled, not run on hardware
//
#include <hip/hip_runtime.h>
#include <hip/hip_bf16.h>

typedef __attribute__((ext_vector_type(16))) __bf16 v16bf;
typedef __attribute__((ext_vector_type(8)))  __bf16 v8bf;
typedef __attribute__((ext_vector_type(8)))  float  v8f;

static constexpr int Vv = 50257, Ss = 1024, Ee = 768, Hh = 12, Ll = 12, Bb = 4;
static constexpr int Dd = 64;
static constexpr int Mtok = Bb * Ss;          // 4096 token rows

// ---------------- WMMA fragment assembly helpers -----------------------------

__device__ inline v16bf frag_from(const __bf16* p, int hi_off) {
  v8bf lo = *(const v8bf*)p;
  v8bf hi = *(const v8bf*)(p + hi_off);
  v16bf f;
#pragma unroll
  for (int i = 0; i < 8; ++i) { f[i] = lo[i]; f[8 + i] = hi[i]; }
  return f;
}

// ---------- GEMM: 64x64 per wave (4x4 WMMA tiles), 8 waves / 256-thr block ---
// A  [M,K] row-major bf16 (activations)
// Wt [N,K] row-major bf16 (weights pre-transposed, pre-converted)
// A-frag: lanes 0-15 row m0+lane  K={k0..k0+7}U{k0+16..k0+23};
//         lanes16-31 row m0+ln-16 K={k0+8..k0+15}U{k0+24..k0+31}
// B-frag: col = lane&15; same K split as A-frag

template <bool GELU, bool RESID, bool OUTBF, bool NGUARD>
__global__ __launch_bounds__(256, 1)
void gemm_kernel(const __bf16* __restrict__ A, const __bf16* __restrict__ Wt,
                 const float* __restrict__ bias, const float* __restrict__ resid,
                 float* __restrict__ Cf, __bf16* __restrict__ Cb,
                 int M, int N, int K) {
  int ntb = (N + 63) >> 6;
  int wid = blockIdx.x * (blockDim.x >> 5) + (threadIdx.x >> 5);
  int bm = wid / ntb, bn = wid % ntb;
  if (bm >= (M >> 6)) return;
  int m0 = bm << 6, n0 = bn << 6;
  int lane = threadIdx.x & 31;

  // per-lane element offsets; k0 added in the subscript (keeps global a-space)
  size_t offA[4], offB[4];
#pragma unroll
  for (int i = 0; i < 4; ++i)
    offA[i] = (size_t)(m0 + 16 * i + (lane & 15)) * K + ((lane >> 4) << 3);
#pragma unroll
  for (int j = 0; j < 4; ++j) {
    int col = n0 + 16 * j + (lane & 15);
    if (NGUARD && col >= N) col = N - 1;   // clamp: junk cols are never stored
    offB[j] = (size_t)col * K + ((lane >> 4) << 3);
  }

  v8f acc[4][4] = {};
  for (int k0 = 0; k0 < K; k0 += 32) {
    v16bf af[4], bfr[4];
#pragma unroll
    for (int i = 0; i < 4; ++i) af[i] = frag_from(A + offA[i] + k0, 16);
#pragma unroll
    for (int j = 0; j < 4; ++j) bfr[j] = frag_from(Wt + offB[j] + k0, 16);
#pragma unroll
    for (int i = 0; i < 4; ++i)
#pragma unroll
      for (int j = 0; j < 4; ++j)
        acc[i][j] = __builtin_amdgcn_wmma_f32_16x16x32_bf16(
            false, af[i], false, bfr[j], (short)0, acc[i][j], false, false);
  }

#pragma unroll
  for (int j = 0; j < 4; ++j) {
    int n = n0 + 16 * j + (lane & 15);
    if (NGUARD && n >= N) continue;
    float bv = bias ? bias[n] : 0.0f;
#pragma unroll
    for (int i = 0; i < 4; ++i) {
#pragma unroll
      for (int r = 0; r < 8; ++r) {
        int m = m0 + 16 * i + ((lane >> 4) << 3) + r;
        float v = acc[i][j][r] + bv;
        if (GELU)  v = 0.5f * v * (1.0f + erff(v * 0.70710678118654752f));
        if (RESID) v += resid[(size_t)m * N + n];
        if (OUTBF) Cb[(size_t)m * N + n] = (__bf16)v;
        else       Cf[(size_t)m * N + n] = v;
      }
    }
  }
}

// ---------------- weight packing -------------------------------------------

// W [K,N] fp32  ->  Wt [N,K] bf16 (transposed); coalesced writes along K
__global__ void pack_transpose_kernel(const float* __restrict__ W, __bf16* __restrict__ Wt,
                                      int K, int N) {
  size_t i = (size_t)blockIdx.x * blockDim.x + threadIdx.x;
  size_t tot = (size_t)K * N;
  if (i >= tot) return;
  int n = (int)(i / K), k = (int)(i % K);
  Wt[i] = (__bf16)W[(size_t)k * N + n];
}

// straight fp32 -> bf16 (wte is already [V,E] = [N,K])
__global__ void convert_bf16_kernel(const float* __restrict__ in, __bf16* __restrict__ out,
                                    size_t n) {
  size_t i = (size_t)blockIdx.x * blockDim.x + threadIdx.x;
  if (i < n) out[i] = (__bf16)in[i];
}

// ---------------- embedding, layernorm, attention, loss ----------------------

__global__ void embed_kernel(const int* __restrict__ idx, const float* __restrict__ wte,
                             const float* __restrict__ wpe, float* __restrict__ x) {
  int row = blockIdx.x;
  int s   = row % Ss;
  const float* te = wte + (size_t)idx[row] * Ee;
  const float* pe = wpe + (size_t)s * Ee;
  float* xr = x + (size_t)row * Ee;
  for (int e = threadIdx.x; e < Ee; e += blockDim.x) xr[e] = te[e] + pe[e];
}

__global__ void layernorm_kernel(const float* __restrict__ x, const float* __restrict__ g,
                                 const float* __restrict__ b, __bf16* __restrict__ out) {
  __shared__ float red[8];
  int row = blockIdx.x, tid = threadIdx.x, lane = tid & 31, wv = tid >> 5;
  const float* xr = x + (size_t)row * Ee;
  float s = 0.0f;
  for (int i = tid; i < Ee; i += 256) s += xr[i];
  for (int o = 16; o > 0; o >>= 1) s += __shfl_xor(s, o, 32);
  if (lane == 0) red[wv] = s;
  __syncthreads();
  float mu = 0.0f;
#pragma unroll
  for (int i = 0; i < 8; ++i) mu += red[i];
  mu /= (float)Ee;
  __syncthreads();
  float vs = 0.0f;
  for (int i = tid; i < Ee; i += 256) { float d = xr[i] - mu; vs += d * d; }
  for (int o = 16; o > 0; o >>= 1) vs += __shfl_xor(vs, o, 32);
  if (lane == 0) red[wv] = vs;
  __syncthreads();
  float var = 0.0f;
#pragma unroll
  for (int i = 0; i < 8; ++i) var += red[i];
  var /= (float)Ee;
  float rstd = rsqrtf(var + 1e-5f);
  __bf16* orow = out + (size_t)row * Ee;
  for (int i = tid; i < Ee; i += 256)
    orow[i] = (__bf16)((xr[i] - mu) * rstd * g[i] + b[i]);
}

// one wave per (b,h,q); scores row staged in LDS; causal softmax; bf16 output
__global__ void attention_kernel(const float* __restrict__ qkv, __bf16* __restrict__ y) {
  __shared__ float sc[8][Ss];
  int wv = threadIdx.x >> 5, lane = threadIdx.x & 31;
  int q = blockIdx.x * 8 + wv, h = blockIdx.y, b = blockIdx.z;
  const float scale = 0.125f;  // 1/sqrt(64)
  const float* qp = qkv + (size_t)(b * Ss + q) * (3 * Ee) + h * Dd;
  float qr[Dd];
#pragma unroll
  for (int d = 0; d < Dd; ++d) qr[d] = qp[d];
  float* srow = sc[wv];
  float mloc = -3.4e38f;
  for (int k = lane; k <= q; k += 32) {
    const float* kp = qkv + (size_t)(b * Ss + k) * (3 * Ee) + Ee + h * Dd;
    float s = 0.0f;
#pragma unroll
    for (int d = 0; d < Dd; ++d) s += qr[d] * kp[d];
    s *= scale;
    srow[k] = s;
    mloc = fmaxf(mloc, s);
  }
  for (int o = 16; o > 0; o >>= 1) mloc = fmaxf(mloc, __shfl_xor(mloc, o, 32));
  float lsum = 0.0f;
  for (int k = lane; k <= q; k += 32) {
    float p = __expf(srow[k] - mloc);
    srow[k] = p;
    lsum += p;
  }
  for (int o = 16; o > 0; o >>= 1) lsum += __shfl_xor(lsum, o, 32);
  float inv = 1.0f / lsum;
  float a0 = 0.0f, a1 = 0.0f;
  int d0 = lane, d1 = lane + 32;
  for (int k = 0; k <= q; ++k) {
    const float* vp = qkv + (size_t)(b * Ss + k) * (3 * Ee) + 2 * Ee + h * Dd;
    float p = srow[k];
    a0 += p * vp[d0];
    a1 += p * vp[d1];
  }
  __bf16* yo = y + (size_t)(b * Ss + q) * Ee + h * Dd;
  yo[d0] = (__bf16)(a0 * inv);
  yo[d1] = (__bf16)(a1 * inv);
}

__global__ void rowloss_kernel(const float* __restrict__ logits, const int* __restrict__ tgt,
                               float* __restrict__ rowloss, int N) {
  __shared__ float red[8];
  int row = blockIdx.x, tid = threadIdx.x, lane = tid & 31, wv = tid >> 5;
  const float* lr = logits + (size_t)row * N;
  float m = -3.4e38f;
  for (int i = tid; i < N; i += 256) m = fmaxf(m, lr[i]);
  for (int o = 16; o > 0; o >>= 1) m = fmaxf(m, __shfl_xor(m, o, 32));
  if (lane == 0) red[wv] = m;
  __syncthreads();
  float mt = red[0];
#pragma unroll
  for (int i = 1; i < 8; ++i) mt = fmaxf(mt, red[i]);
  __syncthreads();
  float s = 0.0f;
  for (int i = tid; i < N; i += 256) s += __expf(lr[i] - mt);
  for (int o = 16; o > 0; o >>= 1) s += __shfl_xor(s, o, 32);
  if (lane == 0) red[wv] = s;
  __syncthreads();
  if (tid == 0) {
    float tot = 0.0f;
#pragma unroll
    for (int i = 0; i < 8; ++i) tot += red[i];
    rowloss[row] = (mt + __logf(tot)) - lr[tgt[row]];
  }
}

__global__ void loss_reduce_kernel(const float* __restrict__ rowloss, float* __restrict__ out,
                                   int n) {
  __shared__ float red[8];
  int tid = threadIdx.x, lane = tid & 31, wv = tid >> 5;
  float s = 0.0f;
  for (int i = tid; i < n; i += 256) s += rowloss[i];
  for (int o = 16; o > 0; o >>= 1) s += __shfl_xor(s, o, 32);
  if (lane == 0) red[wv] = s;
  __syncthreads();
  if (tid == 0) {
    float tot = 0.0f;
#pragma unroll
    for (int i = 0; i < 8; ++i) tot += red[i];
    out[0] = tot / (float)n;
  }
}

// ---------------- launcher ---------------------------------------------------

static inline int gemm_blocks(int M, int N) {
  long long waves = (long long)(M >> 6) * ((N + 63) >> 6);
  return (int)((waves + 7) / 8);       // 8 waves per 256-thread block
}
static inline int ceil_div(long long a, long long b) { return (int)((a + b - 1) / b); }

extern "C" void kernel_launch(void* const* d_in, const int* in_sizes, int n_in,
                              void* d_out, int out_size, void* d_ws, size_t ws_size,
                              hipStream_t stream) {
  const int*   idx    = (const int*)  d_in[0];
  const int*   tgt    = (const int*)  d_in[1];
  const float* wte    = (const float*)d_in[2];
  const float* wpe    = (const float*)d_in[3];
  const float* ln1_g  = (const float*)d_in[4];
  const float* ln1_b  = (const float*)d_in[5];
  const float* attn_w = (const float*)d_in[6];
  const float* attn_b = (const float*)d_in[7];
  const float* proj_w = (const float*)d_in[8];
  const float* proj_b = (const float*)d_in[9];
  const float* ln2_g  = (const float*)d_in[10];
  const float* ln2_b  = (const float*)d_in[11];
  const float* fc_w   = (const float*)d_in[12];
  const float* fc_b   = (const float*)d_in[13];
  const float* fcp_w  = (const float*)d_in[14];
  const float* fcp_b  = (const float*)d_in[15];
  const float* lnf_g  = (const float*)d_in[16];
  const float* lnf_b  = (const float*)d_in[17];

  // ---- workspace layout ----
  float* x    = (float*)d_ws;                          // [4096, 768]   fp32
  float* qkv  = x    + (size_t)Mtok * Ee;              // [4096, 2304]  fp32
  float* rlos = qkv  + (size_t)Mtok * 3 * Ee;          // [4096]        fp32
  __bf16* bp  = (__bf16*)(rlos + Mtok);
  __bf16* h    = bp;                                   // [4096, 768]   bf16
  __bf16* y    = h    + (size_t)Mtok * Ee;             // [4096, 768]   bf16
  __bf16* mid  = y    + (size_t)Mtok * Ee;             // [4096, 3072]  bf16
  __bf16* wteb = mid  + (size_t)Mtok * 4 * Ee;         // [V, 768]      bf16
  __bf16* awt  = wteb + (size_t)Vv * Ee;               // 12x[2304,768] bf16
  __bf16* pwt  = awt  + (size_t)Ll * 3 * Ee * Ee;      // 12x[768,768]
  __bf16* fwt  = pwt  + (size_t)Ll * Ee * Ee;          // 12x[3072,768]
  __bf16* gwt  = fwt  + (size_t)Ll * 4 * Ee * Ee;      // 12x[768,3072]

  float* logits = (float*)d_out;                       // [4096, 50257]
  float* lossp  = logits + (size_t)Mtok * Vv;          // [1]

  dim3 blk(256);

  // ---- pack weights to transposed bf16 (once per launch) ----
  convert_bf16_kernel<<<ceil_div((size_t)Vv * Ee, 256), blk, 0, stream>>>(
      wte, wteb, (size_t)Vv * Ee);
  for (int l = 0; l < Ll; ++l) {
    pack_transpose_kernel<<<ceil_div((size_t)Ee * 3 * Ee, 256), blk, 0, stream>>>(
        attn_w + (size_t)l * Ee * 3 * Ee, awt + (size_t)l * 3 * Ee * Ee, Ee, 3 * Ee);
    pack_transpose_kernel<<<ceil_div((size_t)Ee * Ee, 256), blk, 0, stream>>>(
        proj_w + (size_t)l * Ee * Ee, pwt + (size_t)l * Ee * Ee, Ee, Ee);
    pack_transpose_kernel<<<ceil_div((size_t)Ee * 4 * Ee, 256), blk, 0, stream>>>(
        fc_w + (size_t)l * Ee * 4 * Ee, fwt + (size_t)l * 4 * Ee * Ee, Ee, 4 * Ee);
    pack_transpose_kernel<<<ceil_div((size_t)4 * Ee * Ee, 256), blk, 0, stream>>>(
        fcp_w + (size_t)l * 4 * Ee * Ee, gwt + (size_t)l * 4 * Ee * Ee, 4 * Ee, Ee);
  }

  embed_kernel<<<Mtok, blk, 0, stream>>>(idx, wte, wpe, x);

  for (int l = 0; l < Ll; ++l) {
    const __bf16* aw = awt + (size_t)l * 3 * Ee * Ee;
    const __bf16* pw = pwt + (size_t)l * Ee * Ee;
    const __bf16* fw = fwt + (size_t)l * 4 * Ee * Ee;
    const __bf16* gw = gwt + (size_t)l * 4 * Ee * Ee;
    const float* ab = attn_b + (size_t)l * 3 * Ee;
    const float* pb = proj_b + (size_t)l * Ee;
    const float* fb = fc_b   + (size_t)l * 4 * Ee;
    const float* gb = fcp_b  + (size_t)l * Ee;

    layernorm_kernel<<<Mtok, blk, 0, stream>>>(x, ln1_g + l * Ee, ln1_b + l * Ee, h);
    gemm_kernel<false, false, false, false><<<gemm_blocks(Mtok, 3 * Ee), blk, 0, stream>>>(
        h, aw, ab, nullptr, qkv, nullptr, Mtok, 3 * Ee, Ee);
    attention_kernel<<<dim3(Ss / 8, Hh, Bb), blk, 0, stream>>>(qkv, y);
    gemm_kernel<false, true, false, false><<<gemm_blocks(Mtok, Ee), blk, 0, stream>>>(
        y, pw, pb, x, x, nullptr, Mtok, Ee, Ee);
    layernorm_kernel<<<Mtok, blk, 0, stream>>>(x, ln2_g + l * Ee, ln2_b + l * Ee, h);
    gemm_kernel<true, false, true, false><<<gemm_blocks(Mtok, 4 * Ee), blk, 0, stream>>>(
        h, fw, fb, nullptr, nullptr, mid, Mtok, 4 * Ee, Ee);
    gemm_kernel<false, true, false, false><<<gemm_blocks(Mtok, Ee), blk, 0, stream>>>(
        mid, gw, gb, x, x, nullptr, Mtok, Ee, 4 * Ee);
  }

  layernorm_kernel<<<Mtok, blk, 0, stream>>>(x, lnf_g, lnf_b, h);
  gemm_kernel<false, false, false, true><<<gemm_blocks(Mtok, Vv), blk, 0, stream>>>(
      h, wteb, nullptr, nullptr, logits, nullptr, Mtok, Vv, Ee);
  rowloss_kernel<<<Mtok, blk, 0, stream>>>(logits, tgt, rlos, Vv);
  loss_reduce_kernel<<<1, blk, 0, stream>>>(rlos, lossp, Mtok);
}
